// CrossAttention_15418932593009
// MI455X (gfx1250) — matrix operand order repaired and checked by reading, not compiled, not run on hardware
//
#include <hip/hip_runtime.h>
#include <hip/hip_bf16.h>

typedef __attribute__((ext_vector_type(16))) _Float16 v16h;
typedef __attribute__((ext_vector_type(8)))  _Float16 v8h;
typedef __attribute__((ext_vector_type(2)))  _Float16 v2h;
typedef __attribute__((ext_vector_type(2)))  __fp16   v2fp;
typedef __attribute__((ext_vector_type(8)))  float    v8f;

union frag16 { v16h v; v8h h[2]; v2h p[8]; v2fp q[8]; };

#define HN 8192   // heads * seq = 8 * 1024
#define DH 64     // head dim
#define NB 4      // batch
#define CHUNK 64              // keys per LDS stage
#define ROWB 144              // 128B row + 16B pad (bank-conflict stagger)
#define KBUF (64 * ROWB)      // K chunk: 64 j-rows x 64 d (f16)
#define VBUF (64 * ROWB)      // Vt chunk: 64 d-rows x 64 j (f16)

// fold softmax scale (1/sqrt(64)) and log2(e) into Q so inner loop uses exp2
#define QSCALE 0.18033688011112042f   // 0.125 * 1.4426950408889634

// ---------------------------------------------------------------------------
// Kernel 1: bottlenecked QKV projections (fp32 math, f16 outputs).
// Q,K stored (b, h*n, d) row-major; V stored transposed (b, d, h*n).
// ---------------------------------------------------------------------------
__global__ __launch_bounds__(256) void qkv_kernel(
    const float* __restrict__ x,
    const float* __restrict__ wq1, const float* __restrict__ wq2,
    const float* __restrict__ wk1, const float* __restrict__ wk2,
    const float* __restrict__ wv1, const float* __restrict__ wv2,
    _Float16* __restrict__ Qg, _Float16* __restrict__ Kg,
    _Float16* __restrict__ Vtg)
{
    __shared__ float xs[256];
    __shared__ float t1s[3 * 40];
    const int t   = threadIdx.x;
    const int rid = blockIdx.x;            // 0..4095 : (b, n)
    const int b   = rid >> 10;
    const int nn  = rid & 1023;

    xs[t] = x[(size_t)rid * 256 + t];
    __syncthreads();

    if (t < 120) {
        const int proj = t / 40;           // 0=q 1=k 2=v
        const int jj   = t - proj * 40;
        const float* w1 = (proj == 0) ? wq1 : (proj == 1) ? wk1 : wv1;
        float acc = 0.f;
        #pragma unroll 8
        for (int i = 0; i < 256; ++i) acc += xs[i] * w1[i * 40 + jj];
        if (proj == 1) acc = acc / (1.f + __expf(-acc));   // SiLU on K bottleneck
        t1s[t] = acc;
    }
    __syncthreads();

    #pragma unroll
    for (int k = 0; k < 6; ++k) {
        const int oid  = t + k * 256;      // 0..1535
        const int proj = oid >> 9;
        const int col  = oid & 511;
        const float* w2 = (proj == 0) ? wq2 : (proj == 1) ? wk2 : wv2;
        const float* t1 = t1s + proj * 40;
        float acc = 0.f;
        #pragma unroll
        for (int j = 0; j < 40; ++j) acc += t1[j] * w2[j * 512 + col];
        const int h  = col >> 6;
        const int d  = col & 63;
        const int hn = h * 1024 + nn;      // split(): (b, h, n, d)
        if (proj == 0)
            Qg[((size_t)b * HN + hn) * DH + d] = (_Float16)(acc * QSCALE);
        else if (proj == 1)
            Kg[((size_t)b * HN + hn) * DH + d] = (_Float16)acc;
        else
            Vtg[((size_t)b * DH + d) * HN + hn] = (_Float16)acc;
    }
}

// ---------------------------------------------------------------------------
// Kernel 2: flash attention over j = 0..8191, softmax in base-2.
// One wave = one 16-row Q tile; 8 waves/block share double-buffered 64-key
// K/V chunks via LDS (one barrier + one O-rescale per 16 WMMAs).
// S^T = K*Q^T so the D-layout -> A-layout conversion for P is in-lane.
// ---------------------------------------------------------------------------
__global__ __launch_bounds__(256) void attn_kernel(
    const _Float16* __restrict__ Qg, const _Float16* __restrict__ Kg,
    const _Float16* __restrict__ Vtg, float* __restrict__ Og)
{
    __shared__ __align__(16) unsigned char ldsK[2 * KBUF];
    __shared__ __align__(16) unsigned char ldsV[2 * VBUF];

    const int t    = threadIdx.x;
    const int wave = t >> 5;
    const int lane = t & 31;
    const int hf   = lane >> 4;      // half-wave
    const int li   = lane & 15;
    const int bb   = blockIdx.y;
    const int ibase = blockIdx.x * 128 + wave * 16;

    // Q as B-fragments (B = Q^T, 32x16 over K=d), resident for whole loop.
    frag16 bq0, bq1;
    {
        const _Float16* qrow = Qg + ((size_t)bb * HN + ibase + li) * DH;
        bq0.h[0] = *(const v8h*)(qrow + hf * 16);
        bq0.h[1] = *(const v8h*)(qrow + hf * 16 + 8);
        bq1.h[0] = *(const v8h*)(qrow + 32 + hf * 16);
        bq1.h[1] = *(const v8h*)(qrow + 32 + hf * 16 + 8);
    }

    v8f o[4];
    #pragma unroll
    for (int i = 0; i < 4; ++i) o[i] = (v8f){};
    float m = -3.0e38f, l = 0.f;

    const _Float16* kbase = Kg + (size_t)bb * HN * DH;
    const _Float16* vbase = Vtg + (size_t)bb * DH * HN;
    // both copies: 64 rows x 8 quads(16B); each thread moves quads t and t+256
    const int r0 = t >> 3, s0q = t & 7;          // quad t
    const int r1 = (t + 256) >> 3, s1q = t & 7;  // quad t+256

    // preload chunk 0 into buffer 0
    {
        *(float4*)(ldsK + r0 * ROWB + s0q * 16) =
            *(const float4*)(kbase + (size_t)r0 * DH + s0q * 8);
        *(float4*)(ldsK + r1 * ROWB + s1q * 16) =
            *(const float4*)(kbase + (size_t)r1 * DH + s1q * 8);
        *(float4*)(ldsV + r0 * ROWB + s0q * 16) =
            *(const float4*)(vbase + (size_t)r0 * HN + s0q * 8);
        *(float4*)(ldsV + r1 * ROWB + s1q * 16) =
            *(const float4*)(vbase + (size_t)r1 * HN + s1q * 8);
    }
    __syncthreads();

    for (int jc = 0; jc < HN; jc += CHUNK) {
        const int cur = (jc >> 6) & 1;
        const unsigned char* bK = ldsK + cur * KBUF;
        const unsigned char* bV = ldsV + cur * VBUF;
        const bool has_next = (jc + CHUNK) < HN;

        // prefetch next chunk from global (latency hides under WMMAs)
        float4 nk0, nk1, nv0, nv1;
        if (has_next) {
            const int jn = jc + CHUNK;
            nk0 = *(const float4*)(kbase + (size_t)(jn + r0) * DH + s0q * 8);
            nk1 = *(const float4*)(kbase + (size_t)(jn + r1) * DH + s1q * 8);
            nv0 = *(const float4*)(vbase + (size_t)r0 * HN + jn + s0q * 8);
            nv1 = *(const float4*)(vbase + (size_t)r1 * HN + jn + s1q * 8);
        }

        // ---- QK^T : 4 m-tiles (j), K-dim 64 -> 8 WMMAs --------------------
        v8f s[4];
        const v8f z = (v8f){};
        #pragma unroll
        for (int mt = 0; mt < 4; ++mt) {
            frag16 a0, a1;
            const unsigned char* kr = bK + (mt * 16 + li) * ROWB + hf * 32;
            a0.h[0] = *(const v8h*)(kr);      a0.h[1] = *(const v8h*)(kr + 16);
            a1.h[0] = *(const v8h*)(kr + 64); a1.h[1] = *(const v8h*)(kr + 80);
            s[mt] = __builtin_amdgcn_wmma_f32_16x16x32_f16(false, a0.v, false, bq0.v, (short)0, z,     false, false);
            s[mt] = __builtin_amdgcn_wmma_f32_16x16x32_f16(false, a1.v, false, bq1.v, (short)0, s[mt], false, false);
        }

        // ---- online softmax for this lane's i = li (base-2 domain) --------
        float cmax = s[0][0];
        #pragma unroll
        for (int mt = 0; mt < 4; ++mt)
            #pragma unroll
            for (int r = 0; r < 8; ++r) cmax = fmaxf(cmax, s[mt][r]);
        cmax = fmaxf(cmax, __shfl_xor(cmax, 16, 32));
        const float mnew = fmaxf(m, cmax);

        float p[4][8], rsum = 0.f;
        #pragma unroll
        for (int mt = 0; mt < 4; ++mt)
            #pragma unroll
            for (int r = 0; r < 8; ++r) {
                p[mt][r] = __builtin_amdgcn_exp2f(s[mt][r] - mnew);
                rsum += p[mt][r];
            }
        rsum += __shfl_xor(rsum, 16, 32);
        const float alpha = __builtin_amdgcn_exp2f(m - mnew);
        l = l * alpha + rsum;
        m = mnew;

        // rescale O once per 64 keys: row r+8*hf needs alpha from lane r+8*hf
        #pragma unroll
        for (int r = 0; r < 8; ++r) {
            const float ar = __shfl(alpha, r + 8 * hf, 32);
            o[0][r] *= ar; o[1][r] *= ar; o[2][r] *= ar; o[3][r] *= ar;
        }

        // P as A-fragments (in-lane f32->f16): pa0 = j 0..31, pa1 = j 32..63
        frag16 pa0, pa1;
        #pragma unroll
        for (int e = 0; e < 4; ++e) {
            pa0.q[e]     = __builtin_amdgcn_cvt_pkrtz(p[0][2 * e], p[0][2 * e + 1]);
            pa0.q[e + 4] = __builtin_amdgcn_cvt_pkrtz(p[1][2 * e], p[1][2 * e + 1]);
            pa1.q[e]     = __builtin_amdgcn_cvt_pkrtz(p[2][2 * e], p[2][2 * e + 1]);
            pa1.q[e + 4] = __builtin_amdgcn_cvt_pkrtz(p[3][2 * e], p[3][2 * e + 1]);
        }

        // ---- P*V : 4 d-tiles x 2 j-halves -> 8 WMMAs ----------------------
        #pragma unroll
        for (int dt = 0; dt < 4; ++dt) {
            frag16 bv0, bv1;
            const unsigned char* vr = bV + (dt * 16 + li) * ROWB + hf * 16;
            bv0.h[0] = *(const v8h*)(vr);      bv0.h[1] = *(const v8h*)(vr + 32);
            bv1.h[0] = *(const v8h*)(vr + 64); bv1.h[1] = *(const v8h*)(vr + 96);
            o[dt] = __builtin_amdgcn_wmma_f32_16x16x32_f16(false, pa0.v, false, bv0.v, (short)0, o[dt], false, false);
            o[dt] = __builtin_amdgcn_wmma_f32_16x16x32_f16(false, pa1.v, false, bv1.v, (short)0, o[dt], false, false);
        }

        // publish next chunk to the other buffer
        if (has_next) {
            unsigned char* wK = ldsK + (1 - cur) * KBUF;
            unsigned char* wV = ldsV + (1 - cur) * VBUF;
            *(float4*)(wK + r0 * ROWB + s0q * 16) = nk0;
            *(float4*)(wK + r1 * ROWB + s1q * 16) = nk1;
            *(float4*)(wV + r0 * ROWB + s0q * 16) = nv0;
            *(float4*)(wV + r1 * ROWB + s1q * 16) = nv1;
        }
        __syncthreads();
    }

    // epilogue: divide by l, write O (b, h*n, d) f32
    const float invl = 1.0f / l;
    float* obase = Og + ((size_t)bb * HN + ibase) * DH;
    #pragma unroll
    for (int r = 0; r < 8; ++r) {
        const float wr = __shfl(invl, r + 8 * hf, 32);
        const int row  = r + 8 * hf;
        #pragma unroll
        for (int dt = 0; dt < 4; ++dt)
            obase[(size_t)row * DH + dt * 16 + li] = o[dt][r] * wr;
    }
}

// ---------------------------------------------------------------------------
// Kernel 3: un-split heads, out = O @ wo + bo   (b, n, 256) fp32
// ---------------------------------------------------------------------------
__global__ __launch_bounds__(256) void outproj_kernel(
    const float* __restrict__ Og, const float* __restrict__ wo,
    const float* __restrict__ bo, float* __restrict__ out)
{
    __shared__ float os[512];
    const int t   = threadIdx.x;
    const int rid = blockIdx.x;
    const int b   = rid >> 10;
    const int nn  = rid & 1023;

    #pragma unroll
    for (int k = 0; k < 2; ++k) {
        const int e = t + k * 256;
        const int h = e >> 6, d = e & 63;
        os[e] = Og[(((size_t)b * 8 + h) * 1024 + nn) * DH + d];
    }
    __syncthreads();

    float acc = bo[t];
    #pragma unroll 8
    for (int e = 0; e < 512; ++e) acc += os[e] * wo[(size_t)e * 256 + t];
    out[(size_t)rid * 256 + t] = acc;
}

// ---------------------------------------------------------------------------
extern "C" void kernel_launch(void* const* d_in, const int* in_sizes, int n_in,
                              void* d_out, int out_size, void* d_ws, size_t ws_size,
                              hipStream_t stream) {
    (void)in_sizes; (void)n_in; (void)out_size; (void)ws_size;
    const float* x   = (const float*)d_in[0];
    const float* wq1 = (const float*)d_in[1];
    const float* wq2 = (const float*)d_in[2];
    const float* wk1 = (const float*)d_in[3];
    const float* wk2 = (const float*)d_in[4];
    const float* wv1 = (const float*)d_in[5];
    const float* wv2 = (const float*)d_in[6];
    const float* wo  = (const float*)d_in[7];
    const float* bo  = (const float*)d_in[8];
    float* out = (float*)d_out;

    char* ws = (char*)d_ws;
    _Float16* Qg  = (_Float16*)(ws);                       // 4 MB
    _Float16* Kg  = (_Float16*)(ws + ((size_t)4  << 20));  // 4 MB
    _Float16* Vtg = (_Float16*)(ws + ((size_t)8  << 20));  // 4 MB
    float*    Og  = (float*)   (ws + ((size_t)12 << 20));  // 8 MB

    qkv_kernel<<<NB * 1024, 256, 0, stream>>>(x, wq1, wq2, wk1, wk2, wv1, wv2,
                                              Qg, Kg, Vtg);
    attn_kernel<<<dim3(HN / 128, NB), 256, 0, stream>>>(Qg, Kg, Vtg, Og);
    outproj_kernel<<<NB * 1024, 256, 0, stream>>>(Og, wo, bo, out);
}
